// DPLRSSMBlock_76347338654365
// MI455X (gfx1250) — matrix operand order repaired and checked by reading, not compiled, not run on hardware
//
#include <hip/hip_runtime.h>
#include <math.h>

// Problem dims (from reference): H=1024, RANK=4, BATCH=128, T=256
#define H_DIM   1024
#define RANK_R  4
#define BATCH_B 128
#define T_STEPS 256
#define M_DIM   (BATCH_B * T_STEPS)   // 32768
#define N_DIM   H_DIM                 // 1024
#define K_DIM   H_DIM                 // 1024

// ---- CDNA5 WMMA types ----
typedef __attribute__((ext_vector_type(16))) __bf16          v16bf;
typedef __attribute__((ext_vector_type(8)))  float           v8f;
typedef __attribute__((ext_vector_type(4)))  unsigned short  su4;
typedef __attribute__((ext_vector_type(8)))  unsigned short  su8;
typedef __attribute__((ext_vector_type(16))) unsigned short  su16;

// round-to-nearest-even f32 -> bf16 (bit manipulation; avoids scalar __bf16 math)
__device__ __forceinline__ unsigned short f2bf(float f) {
  unsigned u = __builtin_bit_cast(unsigned, f);
  u += 0x7FFFu + ((u >> 16) & 1u);
  return (unsigned short)(u >> 16);
}

// Single-instruction v_tanh_f32 if the builtin exists on this toolchain
// (CDNA5 ISA lists TANH among TRANS32 ops); otherwise libm fallback.
__device__ __forceinline__ float fast_tanh(float x) {
#if __has_builtin(__builtin_amdgcn_tanhf)
  return __builtin_amdgcn_tanhf(x);
#elif __has_builtin(__builtin_amdgcn_tanh_f32)
  return __builtin_amdgcn_tanh_f32(x);
#else
  return tanhf(x);
#endif
}

// ---------------------------------------------------------------------------
// Prep kernel (runs once per call, tiny): Bt[n][k] = bf16(B[k][n]).
// Classic 32x32 LDS-tiled transpose; coalesced read and coalesced bf16 write.
// ---------------------------------------------------------------------------
__global__ __launch_bounds__(256)
void conv_transpose_B(const float* __restrict__ Bg, unsigned short* __restrict__ Bt) {
  __shared__ float tile[32][33];
  const int nb = blockIdx.x * 32;   // n base
  const int kb = blockIdx.y * 32;   // k base
  const int tx = threadIdx.x;       // 0..31
  const int ty = threadIdx.y;       // 0..7
  #pragma unroll
  for (int i = 0; i < 32; i += 8)
    tile[ty + i][tx] = Bg[(size_t)(kb + ty + i) * N_DIM + nb + tx];  // tile[k][n]
  __syncthreads();
  #pragma unroll
  for (int i = 0; i < 32; i += 8)
    Bt[(size_t)(nb + ty + i) * K_DIM + kb + tx] = f2bf(tile[tx][ty + i]); // Bt[n][k]
}

// ---------------------------------------------------------------------------
// Kernel 1: xB = x @ B   (M=32768, N=K=1024), f32 in / f32 out,
// bf16 WMMA (v_wmma_f32_16x16x32_bf16) with f32 accumulation.
// Block = 256 threads = 8 waves (wave32). Tile 128x128, K-step 32.
// Wave grid 2x4: each wave owns a 64x32 sub-tile = 4x2 WMMA tiles.
// Ping-pong LDS double buffering: one barrier per K-iteration; loads for
// tile k+1 overlap WMMAs of tile k.
// PRET=true: B was pre-transposed+converted to bf16 Bt[N][K] in workspace.
// PRET=false: fallback, transpose B f32 tiles into LDS in-loop.
// ---------------------------------------------------------------------------
constexpr int BM = 128, BN = 128, BK = 32;
constexpr int LDT = BK + 8;   // padded LDS row stride (ushorts): 80B, 16B-aligned rows

template <bool PRET>
__global__ __launch_bounds__(256)
void dplr_xb_gemm(const float* __restrict__ A,             // x  [M,K] row-major f32
                  const float* __restrict__ Bf,            // B  [K,N] row-major f32
                  const unsigned short* __restrict__ Btb,  // Bt [N,K] bf16 (PRET only)
                  float* __restrict__ C)                   // xB [M,N] f32
{
  __shared__ __align__(16) unsigned short As[2][BM][LDT];  // row-major [m][k]
  __shared__ __align__(16) unsigned short Bs[2][BN][LDT];  // transposed [n][k]

  const int tid    = threadIdx.x;
  const int lane   = tid & 31;
  const int wave   = tid >> 5;
  const int waveM  = wave >> 2;        // 0..1
  const int waveN  = wave & 3;         // 0..3
  const int l16    = lane & 15;
  const int hiHalf = (lane >= 16);     // second 16 lanes of the wave
  const int m0 = blockIdx.y * BM;
  const int n0 = blockIdx.x * BN;

  auto loadTiles = [&](int k0, int buf) {
    // --- A tile: 128x32 f32 -> bf16, row-major; 4 float4 per thread ---
    #pragma unroll
    for (int i = 0; i < 4; i++) {
      int slot = tid + i * 256;            // 0..1023 float4 slots
      int row  = slot >> 3;                // 8 float4 per row
      int cf   = (slot & 7) << 2;          // k offset in elements
      float4 f = *(const float4*)(A + (size_t)(m0 + row) * K_DIM + k0 + cf);
      su4 q; q.x = f2bf(f.x); q.y = f2bf(f.y); q.z = f2bf(f.z); q.w = f2bf(f.w);
      *(su4*)&As[buf][row][cf] = q;        // ds_store_b64
    }
    if (PRET) {
      // --- B tile: already bf16 [n][k]; 2x 16B global loads + 2x ds_store_b128 ---
      #pragma unroll
      for (int i = 0; i < 2; i++) {
        int slot = tid + i * 256;          // 0..511 su8 slots
        int row  = slot >> 2;              // 4 su8 per row (32 bf16)
        int cp   = (slot & 3) << 3;        // k offset in elements
        su8 v = *(const su8*)(Btb + (size_t)(n0 + row) * K_DIM + k0 + cp);
        *(su8*)&Bs[buf][row][cp] = v;      // ds_store_b128
      }
    } else {
      // --- fallback: B tile 32x128 f32, coalesced read, transposed b16 stores ---
      #pragma unroll
      for (int i = 0; i < 4; i++) {
        int slot = tid + i * 256;
        int krow = slot >> 5;              // 32 float4 per K-row
        int cf   = (slot & 31) << 2;       // n offset in elements
        float4 f = *(const float4*)(Bf + (size_t)(k0 + krow) * N_DIM + n0 + cf);
        Bs[buf][cf + 0][krow] = f2bf(f.x);
        Bs[buf][cf + 1][krow] = f2bf(f.y);
        Bs[buf][cf + 2][krow] = f2bf(f.z);
        Bs[buf][cf + 3][krow] = f2bf(f.w);
      }
    }
  };

  v8f acc[4][2];
  #pragma unroll
  for (int mt = 0; mt < 4; mt++)
    #pragma unroll
    for (int nt = 0; nt < 2; nt++)
      #pragma unroll
      for (int e = 0; e < 8; e++) acc[mt][nt][e] = 0.0f;

  loadTiles(0, 0);
  __syncthreads();

  int buf = 0;
  for (int k0 = 0; k0 < K_DIM; k0 += BK) {
    if (k0 + BK < K_DIM) loadTiles(k0 + BK, buf ^ 1);  // overlaps WMMAs below

    // --- fragments per documented 16-bit A/B VGPR layouts ---
    // A (16x32, MxK): lanes 0-15 row=l16 hold K {0..7,16..23}; lanes 16-31 K {8..15,24..31}
    v16bf afrag[4];
    #pragma unroll
    for (int mt = 0; mt < 4; mt++) {
      int row = waveM * 64 + mt * 16 + l16;
      int kA  = hiHalf ? 8 : 0;
      su8 lo = *(const su8*)&As[buf][row][kA];        // ds_load_b128
      su8 hi = *(const su8*)&As[buf][row][kA + 16];
      su16 comb = __builtin_shufflevector(lo, hi, 0,1,2,3,4,5,6,7,8,9,10,11,12,13,14,15);
      afrag[mt] = __builtin_bit_cast(v16bf, comb);
    }
    // B (32x16, KxN): lanes 0-15 col=l16 hold K {0..7,16..23}; lanes 16-31 K {8..15,24..31}
    v16bf bfrag[2];
    #pragma unroll
    for (int nt = 0; nt < 2; nt++) {
      int col = waveN * 32 + nt * 16 + l16;
      int kB  = hiHalf ? 8 : 0;
      su8 lo = *(const su8*)&Bs[buf][col][kB];
      su8 hi = *(const su8*)&Bs[buf][col][kB + 16];
      su16 comb = __builtin_shufflevector(lo, hi, 0,1,2,3,4,5,6,7,8,9,10,11,12,13,14,15);
      bfrag[nt] = __builtin_bit_cast(v16bf, comb);
    }

    #pragma unroll
    for (int mt = 0; mt < 4; mt++)
      #pragma unroll
      for (int nt = 0; nt < 2; nt++)
        acc[mt][nt] = __builtin_amdgcn_wmma_f32_16x16x32_bf16(
            false, afrag[mt], false, bfrag[nt], (short)0, acc[mt][nt],
            false, false);

    __syncthreads();   // stores to buf^1 done; reads of buf done
    buf ^= 1;
  }

  // --- writeback: C/D layout => VGPR v holds M=v (lanes 0-15) / M=8+v (lanes 16-31), N=l16
  #pragma unroll
  for (int mt = 0; mt < 4; mt++) {
    int rowBase = m0 + waveM * 64 + mt * 16 + (hiHalf ? 8 : 0);
    #pragma unroll
    for (int nt = 0; nt < 2; nt++) {
      int col = n0 + waveN * 32 + nt * 16 + l16;
      #pragma unroll
      for (int v = 0; v < 8; v++)
        C[(size_t)(rowBase + v) * N_DIM + col] = acc[mt][nt][v];
    }
  }
}

// ---------------------------------------------------------------------------
// Kernel 2: sequential scan, in place over d_out (which holds xB).
//   h_t = tanh(d*h + L@(R@h) + u_t)
// One block per batch element; 256 threads, each owning 4 contiguous h indices.
// Rank-4 reduction: wave32 shfl_xor tree + 8-wave LDS combine.
// Latency cuts on the serial chain:
//   - double-buffered reduction scratch -> ONE barrier per step
//   - u_{t+1} prefetched a step ahead
//   - hardware v_tanh_f32 when available
// ---------------------------------------------------------------------------
__global__ __launch_bounds__(256)
void dplr_scan(const float* __restrict__ h0, const float* __restrict__ dvec,
               const float* __restrict__ Lm, const float* __restrict__ Rm,
               float* __restrict__ out)
{
  const int b    = blockIdx.x;
  const int tid  = threadIdx.x;
  const int i0   = tid << 2;
  const int lane = tid & 31;
  const int wv   = tid >> 5;

  float4 hv = *(const float4*)(h0 + (size_t)b * H_DIM + i0);
  float4 dv = *(const float4*)(dvec + i0);
  // L rows for this thread's 4 h indices: L[i][0..3]
  float4 Lr0 = *(const float4*)(Lm + (size_t)(i0 + 0) * RANK_R);
  float4 Lr1 = *(const float4*)(Lm + (size_t)(i0 + 1) * RANK_R);
  float4 Lr2 = *(const float4*)(Lm + (size_t)(i0 + 2) * RANK_R);
  float4 Lr3 = *(const float4*)(Lm + (size_t)(i0 + 3) * RANK_R);
  // R columns for this thread's 4 h indices: R[j][i0..i0+3]
  float4 Rv0 = *(const float4*)(Rm + 0 * H_DIM + i0);
  float4 Rv1 = *(const float4*)(Rm + 1 * H_DIM + i0);
  float4 Rv2 = *(const float4*)(Rm + 2 * H_DIM + i0);
  float4 Rv3 = *(const float4*)(Rm + 3 * H_DIM + i0);

  // Double-buffered reduction scratch: one barrier per step.
  // Reads of red[p] at step t precede this thread's barrier at step t+1,
  // which precedes any writes to red[p] at step t+2.
  __shared__ float red[2][8][4];
  float* base = out + (size_t)b * T_STEPS * H_DIM;

  float4 u_next = *(const float4*)(base + i0);   // prefetch t=0
  for (int t = 0; t < T_STEPS; t++) {
    const int p = t & 1;
    float4 u = u_next;
    if (t + 1 < T_STEPS)
      u_next = *(const float4*)(base + (size_t)(t + 1) * H_DIM + i0);

    // partial rank-4 products
    float p0 = Rv0.x*hv.x + Rv0.y*hv.y + Rv0.z*hv.z + Rv0.w*hv.w;
    float p1 = Rv1.x*hv.x + Rv1.y*hv.y + Rv1.z*hv.z + Rv1.w*hv.w;
    float p2 = Rv2.x*hv.x + Rv2.y*hv.y + Rv2.z*hv.z + Rv2.w*hv.w;
    float p3 = Rv3.x*hv.x + Rv3.y*hv.y + Rv3.z*hv.z + Rv3.w*hv.w;

    #pragma unroll
    for (int off = 16; off > 0; off >>= 1) {
      p0 += __shfl_xor(p0, off, 32);
      p1 += __shfl_xor(p1, off, 32);
      p2 += __shfl_xor(p2, off, 32);
      p3 += __shfl_xor(p3, off, 32);
    }
    if (lane == 0) {
      red[p][wv][0] = p0; red[p][wv][1] = p1;
      red[p][wv][2] = p2; red[p][wv][3] = p3;
    }
    __syncthreads();                       // single barrier per step
    float r0 = 0.f, r1 = 0.f, r2 = 0.f, r3 = 0.f;
    #pragma unroll
    for (int w = 0; w < 8; w++) {
      r0 += red[p][w][0]; r1 += red[p][w][1];
      r2 += red[p][w][2]; r3 += red[p][w][3];
    }

    float4 nh;
    nh.x = fast_tanh(dv.x*hv.x + (Lr0.x*r0 + Lr0.y*r1 + Lr0.z*r2 + Lr0.w*r3) + u.x);
    nh.y = fast_tanh(dv.y*hv.y + (Lr1.x*r0 + Lr1.y*r1 + Lr1.z*r2 + Lr1.w*r3) + u.y);
    nh.z = fast_tanh(dv.z*hv.z + (Lr2.x*r0 + Lr2.y*r1 + Lr2.z*r2 + Lr2.w*r3) + u.z);
    nh.w = fast_tanh(dv.w*hv.w + (Lr3.x*r0 + Lr3.y*r1 + Lr3.z*r2 + Lr3.w*r3) + u.w);
    hv = nh;
    *(float4*)(base + (size_t)t * H_DIM + i0) = hv;   // overwrite u_t with h_t
  }
}

// ---------------------------------------------------------------------------
extern "C" void kernel_launch(void* const* d_in, const int* in_sizes, int n_in,
                              void* d_out, int out_size, void* d_ws, size_t ws_size,
                              hipStream_t stream) {
  (void)in_sizes; (void)n_in; (void)out_size;
  const float* x  = (const float*)d_in[0];   // [BATCH,T,H]
  const float* h0 = (const float*)d_in[1];   // [BATCH,H]
  const float* dv = (const float*)d_in[2];   // [H]
  const float* L  = (const float*)d_in[3];   // [H,RANK]
  const float* R  = (const float*)d_in[4];   // [RANK,H]
  const float* B  = (const float*)d_in[5];   // [H,H]
  float* out = (float*)d_out;                // [BATCH,T,H]

  const size_t btBytes = (size_t)N_DIM * K_DIM * sizeof(unsigned short); // 2 MB
  dim3 grid(N_DIM / BN, M_DIM / BM);         // (8, 256)

  if (ws_size >= btBytes) {
    unsigned short* Bt = (unsigned short*)d_ws;
    conv_transpose_B<<<dim3(N_DIM / 32, K_DIM / 32), dim3(32, 8), 0, stream>>>(B, Bt);
    dplr_xb_gemm<true><<<grid, 256, 0, stream>>>(x, B, Bt, out);
  } else {
    dplr_xb_gemm<false><<<grid, 256, 0, stream>>>(x, B, nullptr, out);
  }
  dplr_scan<<<BATCH_B, 256, 0, stream>>>(h0, dv, L, R, out); // out <- scan (in place)
}
